// SelfAttention_30648886624564
// MI455X (gfx1250) — matrix operand order repaired and checked by reading, not compiled, not run on hardware
//
#include <hip/hip_runtime.h>
#include <hip/hip_bf16.h>

typedef __attribute__((ext_vector_type(16))) __bf16 v16bf;
typedef __attribute__((ext_vector_type(8)))  float  v8f;
typedef __attribute__((ext_vector_type(8)))  unsigned int v8u;
typedef __attribute__((ext_vector_type(4)))  unsigned int u32x4;
typedef __attribute__((ext_vector_type(8)))  unsigned int u32x8;
typedef unsigned short ushort_t;
typedef unsigned int   uint_t;

#define DEV static __device__ __forceinline__

// ---------- constants ----------
#define BATCH  4
#define SEQ    2048
#define DMODEL 1024
#define DHEAD  128
#define ROWS   (BATCH * SEQ)     // 8192
#define BM     128               // query rows per block
#define KT     64                // keys per attention tile
#define KSPLIT 4                 // flash-decoding key splits
#define KCHUNK (SEQ / KSPLIT)    // 512 keys per block

// ---------- helpers ----------
DEV uint_t f2bf_bits(float f) {
    uint_t u = __builtin_bit_cast(uint_t, f);
    uint_t r = u + 0x7FFFu + ((u >> 16) & 1u);   // RNE-ish bf16
    return r >> 16;
}
DEV ushort_t f2bf(float f) { return (ushort_t)f2bf_bits(f); }

DEV v8f wmma_bf16(v16bf a, v16bf b, v8f c) {
    return __builtin_amdgcn_wmma_f32_16x16x32_bf16(
        false, a, false, b, (short)0, c, false, false);
}

// A-operand (16x32 bf16): lane%16 = M; VGPR p -> K-pair (p&3)+(hi?4:0)+(p&4?8:0)
DEV v16bf load_frag_A(const ushort_t* base, int stride, int rowBase, int wordOff, int lane) {
    const int m  = lane & 15;
    const int hi = (lane >> 4) & 1;
    const ushort_t* rp = base + (size_t)(rowBase + m) * stride;
    v8u t;
#pragma unroll
    for (int p = 0; p < 8; ++p) {
        int w = (p & 3) + (hi ? 4 : 0) + ((p & 4) ? 8 : 0);
        t[p] = *(const uint_t*)(rp + 2 * (wordOff + w));
    }
    return __builtin_bit_cast(v16bf, t);
}

// B-operand (32x16 bf16): lane%16 = N; VGPR j -> K-pair (j&3)+(hi?4:0)+(j&4?8:0)
DEV v16bf load_frag_B(const ushort_t* base, int stride, int nBase, int wordOff, int lane) {
    const int n  = nBase + (lane & 15);
    const int hi = (lane >> 4) & 1;
    const ushort_t* rp = base + (size_t)n * stride;
    v8u t;
#pragma unroll
    for (int j = 0; j < 8; ++j) {
        int w = wordOff + (j & 3) + (hi ? 4 : 0) + ((j & 4) ? 8 : 0);
        t[j] = *(const uint_t*)(rp + 2 * w);
    }
    return __builtin_bit_cast(v16bf, t);
}

// ---------- Tensor Data Mover: 2-D bf16 tile (64 rows x 128 cols) -> LDS ----------
// LDS gets padded rows: after every 64 DWORDs written, skip 1 DWORD
// (pad_interval code 5 = 64 DW, pad_amount code 0 = 1 DW) => 130-short row stride.
DEV void tdm_load_k_tile(const ushort_t* gsrc, uint_t lds_byte_off) {
    unsigned long long ga = (unsigned long long)(size_t)gsrc;
    u32x4 g0;
    g0[0] = 1u;                                  // count=1, user descriptor
    g0[1] = lds_byte_off;                        // lds_addr [63:32]
    g0[2] = (uint_t)ga;                          // global_addr lo
    g0[3] = (uint_t)(ga >> 32) | (2u << 30);     // global_addr hi | type=2 (image)
    u32x8 g1;
    g1[0] = (1u << 16) | (1u << 20) | (5u << 22); // data_size=2B, pad_enable, pad_interval=64DW, pad_amount=1DW
    g1[1] = (uint_t)DHEAD << 16;                  // tensor_dim0 = 128   (bits 79:48)
    g1[2] = (uint_t)SEQ << 16;                    // tensor_dim1 = 2048  (bits 111:80)
    g1[3] = (uint_t)DHEAD << 16;                  // tile_dim0  = 128    (bits 127:112)
    g1[4] = (uint_t)KT;                           // tile_dim1  = 64, tile_dim2 = 0
    g1[5] = (uint_t)DHEAD;                        // tensor_dim0_stride = 128 (bits 191:160)
    g1[6] = 0u;
    g1[7] = 0u;
    asm volatile("tensor_load_to_lds %0, %1" :: "s"(g0), "s"(g1) : "memory");
}

// =====================================================================
// Kernel 1: QKV projection.  C[128x128] = X[128x1024] * W[1024x128].
// Single-buffer LDS staging: 8 waves of TLP hide the staging latency and
// the low register pressure lets the compiler hoist all fragment addresses
// and overlap ds_loads with WMMAs (partial s_wait_dscnt).
// grid = (ROWS/BM, 3), block = 256
// =====================================================================
__global__ __launch_bounds__(256)
void qkv_proj_kernel(const float* __restrict__ X,
                     const float* __restrict__ Wq,
                     const float* __restrict__ Wk,
                     const float* __restrict__ Wv,
                     ushort_t* __restrict__ Qb,
                     ushort_t* __restrict__ Kb,
                     ushort_t* __restrict__ Vb)
{
    const int wy = blockIdx.y;
    const float* __restrict__ W = (wy == 0) ? Wq : (wy == 1) ? Wk : Wv;
    ushort_t* __restrict__ Out  = (wy == 0) ? Qb : (wy == 1) ? Kb : Vb;

    const int row0 = blockIdx.x * BM;
    const int tid  = threadIdx.x;
    const int wave = tid >> 5;
    const int lane = tid & 31;
    const int hi   = (lane >> 4) & 1;

    __shared__ ushort_t Xs[BM * 34];   // 128 rows x 32 kc (+2 pad), bf16
    __shared__ ushort_t Ws[128 * 34];  // 128 n   x 32 kc (+2 pad), bf16 (transposed W)

    const v8f vzero = {0.f,0.f,0.f,0.f,0.f,0.f,0.f,0.f};
    v8f acc[8];
#pragma unroll
    for (int t = 0; t < 8; ++t) acc[t] = vzero;

    for (int k0 = 0; k0 < DMODEL; k0 += 32) {
        __syncthreads();
        // stage X tile 128x32 (coalesced rows of 32 floats)
#pragma unroll
        for (int i = 0; i < 16; ++i) {
            int e = tid + 256 * i;          // 4096 elements
            int r = e >> 5, c = e & 31;
            Xs[r * 34 + c] = f2bf(X[(size_t)(row0 + r) * DMODEL + k0 + c]);
        }
        // stage W tile 32x128 transposed -> Ws[n][kc]
#pragma unroll
        for (int i = 0; i < 16; ++i) {
            int e = tid + 256 * i;          // 4096 elements
            int kc = e >> 7, n = e & 127;
            Ws[n * 34 + kc] = f2bf(W[(size_t)(k0 + kc) * DHEAD + n]);
        }
        __syncthreads();

        v16bf a = load_frag_A(Xs, 34, wave * 16, 0, lane);
#pragma unroll
        for (int t = 0; t < 8; ++t) {
            v16bf b = load_frag_B(Ws, 34, t * 16, 0, lane);
            acc[t] = wmma_bf16(a, b, acc[t]);
        }
    }

    // store bf16 results: row = row0 + wave*16 + g + 8*hi, col = t*16 + lane%16
#pragma unroll
    for (int t = 0; t < 8; ++t) {
#pragma unroll
        for (int g = 0; g < 8; ++g) {
            int r = row0 + wave * 16 + g + 8 * hi;
            int c = t * 16 + (lane & 15);
            Out[(size_t)r * DHEAD + c] = f2bf(acc[t][g]);
        }
    }
}

// =====================================================================
// Kernel 2: flash attention partials (split-K flash decoding).
// grid = (SEQ/BM, BATCH, KSPLIT), block = 256.  Each wave owns 16 q rows.
// K tiles arrive via the Tensor Data Mover (async DMA, TENSORcnt).
// Emits unnormalized O plus per-row (m, l) per split.
// =====================================================================
__global__ __launch_bounds__(256)
void attn_partial_kernel(const ushort_t* __restrict__ Qb,
                         const ushort_t* __restrict__ Kb,
                         const ushort_t* __restrict__ Vb,
                         float* __restrict__ Opart,
                         float* __restrict__ Mlpart)
{
    const int b     = blockIdx.y;
    const int q0    = blockIdx.x * BM;
    const int split = blockIdx.z;
    const int tid   = threadIdx.x;
    const int wave  = tid >> 5;
    const int lane  = tid & 31;
    const int hi    = (lane >> 4) & 1;
    const size_t base = (size_t)b * SEQ * DHEAD;

    __shared__ ushort_t Ks[KT * 130];   // 64 keys x 128 dk (+pad via TDM)
    __shared__ ushort_t Vt[128 * 66];   // 128 dv  x 64 keys (+2 pad), V transposed
    __shared__ ushort_t Ps[128 * 66];   // 128 q   x 64 keys (+2 pad), wave-private rows

    v16bf qa[4];
#pragma unroll
    for (int ks = 0; ks < 4; ++ks)
        qa[ks] = load_frag_A(Qb + base + (size_t)q0 * DHEAD, DHEAD, wave * 16, ks * 16, lane);

    const v8f vzero = {0.f,0.f,0.f,0.f,0.f,0.f,0.f,0.f};
    v8f o[8];
    float m[8], l[8];
#pragma unroll
    for (int t = 0; t < 8; ++t) o[t] = vzero;
#pragma unroll
    for (int g = 0; g < 8; ++g) { m[g] = -3.0e38f; l[g] = 0.f; }

    const float sc = 0.088388347648318447f;   // 1/sqrt(128)
    const uint_t ks_lds = (uint_t)(size_t)(void*)&Ks[0];

    for (int kt = 0; kt < KCHUNK / KT; ++kt) {
        const int kv0 = split * KCHUNK + kt * KT;
        __syncthreads();   // everyone done reading previous Ks/Vt

        // K tile via Tensor Data Mover, issued by wave 0
        if (wave == 0)
            tdm_load_k_tile(Kb + base + (size_t)kv0 * DHEAD, ks_lds);

        // V tile transposed by all threads (overlaps the TDM transfer)
#pragma unroll
        for (int i = 0; i < 32; ++i) {
            int e = tid + 256 * i;              // 8192 elements
            int key = e >> 7, dv = e & 127;
            Vt[dv * 66 + key] = Vb[base + (size_t)(kv0 + key) * DHEAD + dv];
        }
        if (kt + 1 < KCHUNK / KT)
            __builtin_prefetch(Vb + base + (size_t)(kv0 + KT) * DHEAD + tid * 32, 0, 1);

        if (wave == 0) __builtin_amdgcn_s_wait_tensorcnt(0);
        __syncthreads();

        // ---- S = Q * K^T ----
        v8f s[4];
#pragma unroll
        for (int t = 0; t < 4; ++t) s[t] = vzero;
#pragma unroll
        for (int ks = 0; ks < 4; ++ks) {
#pragma unroll
            for (int t = 0; t < 4; ++t) {
                v16bf bb = load_frag_B(Ks, 130, t * 16, ks * 16, lane);
                s[t] = wmma_bf16(qa[ks], bb, s[t]);
            }
        }

        // ---- online softmax (max tracked in unscaled domain) ----
#pragma unroll
        for (int g = 0; g < 8; ++g) {
            float pm = -3.0e38f;
#pragma unroll
            for (int t = 0; t < 4; ++t) pm = fmaxf(pm, s[t][g]);
#pragma unroll
            for (int off = 1; off < 16; off <<= 1)
                pm = fmaxf(pm, __shfl_xor(pm, off, 32));

            float newm = fmaxf(m[g], pm);
            float corr = __expf((m[g] - newm) * sc);
            m[g] = newm;
            l[g] *= corr;
#pragma unroll
            for (int t = 0; t < 8; ++t) o[t][g] *= corr;

            float rs = 0.f;
            const int prow = (wave * 16 + g + 8 * hi) * 66;
#pragma unroll
            for (int t = 0; t < 4; ++t) {
                float p = __expf((s[t][g] - newm) * sc);
                rs += p;
                Ps[prow + (lane & 15) + 16 * t] = f2bf(p);
            }
#pragma unroll
            for (int off = 1; off < 16; off <<= 1)
                rs += __shfl_xor(rs, off, 32);
            l[g] += rs;
        }

        // ---- O += P * V ----
#pragma unroll
        for (int ks2 = 0; ks2 < 2; ++ks2) {
            v16bf pa = load_frag_A(Ps, 66, wave * 16, ks2 * 16, lane);
#pragma unroll
            for (int t = 0; t < 8; ++t) {
                v16bf bb = load_frag_B(Vt, 66, t * 16, ks2 * 16, lane);
                o[t] = wmma_bf16(pa, bb, o[t]);
            }
        }
    }

    // ---- emit unnormalized partials + per-row stats ----
#pragma unroll
    for (int g = 0; g < 8; ++g) {
        int q    = q0 + wave * 16 + g + 8 * hi;
        size_t rg = (size_t)split * ROWS + (size_t)b * SEQ + q;
        if ((lane & 15) == 0) {
            Mlpart[rg * 2 + 0] = m[g] * sc;   // store in scaled domain
            Mlpart[rg * 2 + 1] = l[g];
        }
#pragma unroll
        for (int t = 0; t < 8; ++t) {
            int dv = t * 16 + (lane & 15);
            Opart[rg * DHEAD + dv] = o[t][g];
        }
    }
}

// =====================================================================
// Kernel 3: combine split-K partials.  grid = ROWS*DHEAD/256, block = 256
// =====================================================================
__global__ __launch_bounds__(256)
void attn_combine_kernel(const float* __restrict__ Opart,
                         const float* __restrict__ Mlpart,
                         float* __restrict__ Out)
{
    const int idx = blockIdx.x * 256 + threadIdx.x;
    const int row = idx >> 7;
    const int dv  = idx & 127;

    float M = -3.0e38f;
#pragma unroll
    for (int s = 0; s < KSPLIT; ++s)
        M = fmaxf(M, Mlpart[((size_t)s * ROWS + row) * 2 + 0]);

    float L = 0.f, acc = 0.f;
#pragma unroll
    for (int s = 0; s < KSPLIT; ++s) {
        size_t rg = (size_t)s * ROWS + row;
        float w = __expf(Mlpart[rg * 2 + 0] - M);
        acc += w * Opart[rg * DHEAD + dv];
        L   += w * Mlpart[rg * 2 + 1];
    }
    Out[(size_t)row * DHEAD + dv] = acc / L;
}

// =====================================================================
extern "C" void kernel_launch(void* const* d_in, const int* in_sizes, int n_in,
                              void* d_out, int out_size, void* d_ws, size_t ws_size,
                              hipStream_t stream)
{
    (void)in_sizes; (void)n_in; (void)out_size; (void)ws_size;
    const float* X  = (const float*)d_in[0];
    const float* Wq = (const float*)d_in[1];
    const float* Wk = (const float*)d_in[2];
    const float* Wv = (const float*)d_in[3];

    ushort_t* Qb = (ushort_t*)d_ws;                       // 8192x128 bf16 each
    ushort_t* Kb = Qb + (size_t)ROWS * DHEAD;
    ushort_t* Vb = Kb + (size_t)ROWS * DHEAD;
    float* Opart = (float*)(Vb + (size_t)ROWS * DHEAD);   // KSPLIT x ROWS x 128
    float* Mlpart = Opart + (size_t)KSPLIT * ROWS * DHEAD; // KSPLIT x ROWS x 2

    qkv_proj_kernel<<<dim3(ROWS / BM, 3), 256, 0, stream>>>(X, Wq, Wk, Wv, Qb, Kb, Vb);
    attn_partial_kernel<<<dim3(SEQ / BM, BATCH, KSPLIT), 256, 0, stream>>>(
        Qb, Kb, Vb, Opart, Mlpart);
    attn_combine_kernel<<<dim3(ROWS * DHEAD / 256), 256, 0, stream>>>(
        Opart, Mlpart, (float*)d_out);
}
